// SingleStreamBlock_81999515615975
// MI455X (gfx1250) — matrix operand order repaired and checked by reading, not compiled, not run on hardware
//
#include <hip/hip_runtime.h>
#include <math.h>
#include <stdint.h>

// ---------------- problem constants ----------------
#define L_    2048
#define HID_  3072
#define NH_   24
#define HD_   128
#define MLP_  12288
#define NQKV_ (3 * HID_)          // 9216
#define N1_   (3 * HID_ + MLP_)   // 21504
#define NCAT_ (HID_ + MLP_)       // 15360
#define EPS_  1e-6f

typedef __bf16 bf16;
typedef __attribute__((ext_vector_type(16))) __bf16 v16bf;
typedef __attribute__((ext_vector_type(8)))  __bf16 v8bf;
typedef __attribute__((ext_vector_type(8)))  float  v8f;
typedef __attribute__((ext_vector_type(4)))  int    v4i;

union Frag16 { v16bf v; v8bf h[2]; };

// ---- optional gfx1250 async global->LDS path (ASYNCcnt) --------------------
#if defined(__has_builtin)
#  if __has_builtin(__builtin_amdgcn_global_load_async_to_lds_b128)
#    define USE_ASYNC_LDS 1
#  endif
#endif
#ifndef USE_ASYNC_LDS
#  define USE_ASYNC_LDS 0
#endif

#if USE_ASYNC_LDS
typedef __attribute__((address_space(1))) v4i  g_v4i;
typedef __attribute__((address_space(3))) v4i  l_v4i;
typedef __attribute__((address_space(1))) void g_void;
typedef __attribute__((address_space(3))) void l_void;
#endif

static __device__ __forceinline__ void copy16_lds(const bf16* src, bf16* dst_lds) {
#if USE_ASYNC_LDS
    __builtin_amdgcn_global_load_async_to_lds_b128(
        (g_v4i*)(g_void*)src,
        (l_v4i*)(l_void*)dst_lds, 0, 0);
#else
    *(v8bf*)dst_lds = *(const v8bf*)src;
#endif
}

static __device__ __forceinline__ void async_join() {
#if USE_ASYNC_LDS
#  if __has_builtin(__builtin_amdgcn_s_wait_asynccnt)
    __builtin_amdgcn_s_wait_asynccnt(0);
#  else
    asm volatile("s_wait_asynccnt 0x0" ::: "memory");
#  endif
#endif
}

static __device__ __forceinline__ v8f wmma_bf16(const Frag16& a, const Frag16& b, v8f c) {
    return __builtin_amdgcn_wmma_f32_16x16x32_bf16(false, a.v, false, b.v,
                                                   (short)0, c, false, false);
}

static __device__ __forceinline__ uint32_t pack_bf16(float lo, float hi) {
    union { uint32_t u; __bf16 b[2]; } p;
    p.b[0] = (__bf16)lo; p.b[1] = (__bf16)hi;
    return p.u;
}

static __device__ __forceinline__ float gelu_tanh(float v) {
    float u = 0.7978845608028654f * (v + 0.044715f * v * v * v);
    return 0.5f * v * (1.0f + tanhf(u));
}

// ================= kernel 1: m = silu(vec) @ mod_w + mod_b ==================
__global__ __launch_bounds__(256) void ssb_mod_kernel(
    const float* __restrict__ vec, const float* __restrict__ mod_w,
    const float* __restrict__ mod_b, float* __restrict__ m_vec)
{
    __shared__ float sv[HID_];
    for (int i = threadIdx.x; i < HID_; i += 256) {
        float v = vec[i];
        sv[i] = v / (1.0f + __expf(-v));
    }
    __syncthreads();
    int col = blockIdx.x * 256 + threadIdx.x;   // 0..9215
    float acc = mod_b[col];
    for (int k = 0; k < HID_; ++k)
        acc = fmaf(sv[k], mod_w[(size_t)k * NQKV_ + col], acc);
    m_vec[col] = acc;
}

// ====== kernel 2: x_mod = (1+scale)*LN(x) + shift  -> bf16 ==================
__global__ __launch_bounds__(256) void ssb_lnmod_kernel(
    const float* __restrict__ x, const float* __restrict__ m_vec,
    bf16* __restrict__ xmod)
{
    const int row = blockIdx.x;
    const float* xr = x + (size_t)row * HID_;
    float s = 0.f, s2 = 0.f;
    for (int i = threadIdx.x; i < HID_; i += 256) {
        float v = xr[i]; s += v; s2 += v * v;
    }
    for (int m = 16; m >= 1; m >>= 1) { s += __shfl_xor(s, m); s2 += __shfl_xor(s2, m); }
    __shared__ float red[20];
    int wid = threadIdx.x >> 5, lane = threadIdx.x & 31;
    if (lane == 0) { red[wid] = s; red[8 + wid] = s2; }
    __syncthreads();
    if (threadIdx.x == 0) {
        float a = 0.f, b = 0.f;
        for (int w = 0; w < 8; ++w) { a += red[w]; b += red[8 + w]; }
        red[16] = a * (1.0f / HID_);
        red[17] = b * (1.0f / HID_);
    }
    __syncthreads();
    float mu = red[16];
    float var = red[17] - mu * mu;
    float r = rsqrtf(var + EPS_);
    for (int i = threadIdx.x; i < HID_; i += 256) {
        float ln = (xr[i] - mu) * r;
        float v = fmaf(1.0f + m_vec[HID_ + i], ln, m_vec[i]);
        xmod[(size_t)row * HID_ + i] = (bf16)v;
    }
}

// ================= WMMA GEMM core (bf16 A, fp32->bf16 B) ====================
// Block tile 128(M) x 64(N) x 32(K), double-buffered LDS; 8 waves, each wave
// a 32x32 patch (2x2 WMMA). Grid: x = M blocks (16), y = N blocks, so the 16
// M-blocks sharing a weight column-panel run concurrently and hit L2.
// MODE 0: gemm1 epilogue (split qkv / gelu->cat). MODE 1: gemm2 (gate+residual).
#define BM 128
#define BN 64
#define BK 32

template <int MODE>
__global__ __launch_bounds__(256) void ssb_gemm_kernel(
    const bf16* __restrict__ A, const float* __restrict__ Bw,
    const float* __restrict__ bias, int K, int N,
    bf16* __restrict__ qkv_out, bf16* __restrict__ cat_out,
    const float* __restrict__ x, const float* __restrict__ m_vec,
    float* __restrict__ out)
{
    __shared__ bf16 As[2][BM][BK];   // 16 KB
    __shared__ bf16 Bs[2][BN][40];   // 10 KB (row stride 40 -> 80B, 16B aligned)

    const int tid  = threadIdx.x;
    const int wid  = tid >> 5;
    const int lane = tid & 31;
    const int half = lane >> 4;
    const int l15  = lane & 15;
    const int wm   = wid & 3;     // 4 wave rows
    const int wn   = wid >> 2;    // 2 wave cols
    const int m0   = blockIdx.x * BM;
    const int n0   = blockIdx.y * BN;

    auto loadA = [&](int buf, int k0) {
        int r = tid >> 1;
        int c = (tid & 1) * 16;
        const bf16* src = A + (size_t)(m0 + r) * K + k0 + c;
        copy16_lds(src,     &As[buf][r][c]);
        copy16_lds(src + 8, &As[buf][r][c + 8]);
    };
    auto loadB = [&](int buf, int k0) {
        int n  = tid & 63;
        int kq = tid >> 6;                        // 0..3 -> even-k base 2*kq
        const float* bp = Bw + (size_t)(k0 + 2 * kq) * N + n0 + n;
        if (k0 + 2 * BK < K)
            __builtin_prefetch(bp + (size_t)2 * BK * N, 0, 1);
        for (int p = 0; p < 4; ++p) {
            float lo = bp[(size_t)(8 * p) * N];
            float hi = bp[(size_t)(8 * p + 1) * N];
            *(uint32_t*)&Bs[buf][n][2 * kq + 8 * p] = pack_bf16(lo, hi);
        }
    };

    v8f acc[2][2] = {};

    loadA(0, 0);
    loadB(0, 0);
    async_join();
    __syncthreads();

    int buf = 0;
    for (int k0 = 0; k0 < K; k0 += BK, buf ^= 1) {
        if (k0 + BK < K) {            // prefetch next tile into other buffer
            loadA(buf ^ 1, k0 + BK);
            loadB(buf ^ 1, k0 + BK);
        }
        for (int tm = 0; tm < 2; ++tm) {
            Frag16 a;
            int row = wm * 32 + tm * 16 + l15;
            a.h[0] = *(const v8bf*)&As[buf][row][8 * half];
            a.h[1] = *(const v8bf*)&As[buf][row][16 + 8 * half];
            for (int tn = 0; tn < 2; ++tn) {
                Frag16 b;
                int col = wn * 32 + tn * 16 + l15;
                b.h[0] = *(const v8bf*)&Bs[buf][col][8 * half];
                b.h[1] = *(const v8bf*)&Bs[buf][col][16 + 8 * half];
                acc[tm][tn] = wmma_bf16(a, b, acc[tm][tn]);
            }
        }
        async_join();
        __syncthreads();
    }

    // ---- epilogue: C elem (m,n): VGPR r -> m = r + 8*half, n = l15
    for (int tm = 0; tm < 2; ++tm)
        for (int tn = 0; tn < 2; ++tn) {
            int gcol = n0 + wn * 32 + tn * 16 + l15;
            float bcol = bias[gcol];
            for (int r = 0; r < 8; ++r) {
                int grow = m0 + wm * 32 + tm * 16 + r + 8 * half;
                float v = acc[tm][tn][r] + bcol;
                if constexpr (MODE == 0) {
                    if (gcol < NQKV_)
                        qkv_out[(size_t)grow * NQKV_ + gcol] = (bf16)v;
                    else
                        cat_out[(size_t)grow * NCAT_ + HID_ + (gcol - NQKV_)] =
                            (bf16)gelu_tanh(v);
                } else {
                    size_t idx = (size_t)grow * HID_ + gcol;
                    out[idx] = fmaf(m_vec[2 * HID_ + gcol], v, x[idx]);
                }
            }
        }
}

// ============ kernel 4: q/k RMSNorm + scale + RoPE (wave per row) ===========
__global__ __launch_bounds__(256) void ssb_qkprep_kernel(
    const bf16* __restrict__ qkv, const float* __restrict__ pe,
    const float* __restrict__ q_scale, const float* __restrict__ k_scale,
    bf16* __restrict__ qbuf, bf16* __restrict__ kbuf)
{
    const int wid  = threadIdx.x >> 5;
    const int lane = threadIdx.x & 31;
    const int id   = blockIdx.x * 8 + wid;       // 0 .. NH*L-1
    const int h    = id / L_;
    const int l    = id % L_;
    const int d0   = lane * 4;

    const bf16* qs = qkv + (size_t)l * NQKV_ + h * HD_ + d0;
    const bf16* ks = qkv + (size_t)l * NQKV_ + HID_ + h * HD_ + d0;

    float qv[4], kv[4], sq = 0.f, sk = 0.f;
    for (int i = 0; i < 4; ++i) {
        qv[i] = (float)qs[i]; kv[i] = (float)ks[i];
        sq = fmaf(qv[i], qv[i], sq); sk = fmaf(kv[i], kv[i], sk);
    }
    for (int m = 1; m < 32; m <<= 1) { sq += __shfl_xor(sq, m); sk += __shfl_xor(sk, m); }
    float rq = rsqrtf(sq * (1.0f / HD_) + EPS_);
    float rk = rsqrtf(sk * (1.0f / HD_) + EPS_);
    for (int i = 0; i < 4; ++i) {
        qv[i] = qv[i] * rq * q_scale[d0 + i];
        kv[i] = kv[i] * rk * k_scale[d0 + i];
    }
    float qo[4], ko[4];
    for (int p = 0; p < 2; ++p) {
        int i = (d0 >> 1) + p;                          // rotary pair index
        const float* pp = pe + (size_t)l * 256 + i * 4; // [L][64][2][2]
        qo[2 * p]     = pp[0] * qv[2 * p] + pp[1] * qv[2 * p + 1];
        qo[2 * p + 1] = pp[2] * qv[2 * p] + pp[3] * qv[2 * p + 1];
        ko[2 * p]     = pp[0] * kv[2 * p] + pp[1] * kv[2 * p + 1];
        ko[2 * p + 1] = pp[2] * kv[2 * p] + pp[3] * kv[2 * p + 1];
    }
    bf16* qd = qbuf + ((size_t)h * L_ + l) * HD_ + d0;
    bf16* kd = kbuf + ((size_t)h * L_ + l) * HD_ + d0;
    for (int i = 0; i < 4; ++i) { qd[i] = (bf16)qo[i]; kd[i] = (bf16)ko[i]; }
}

// ============ kernel 5: flash attention, WMMA QK^T and PV ===================
#define QT 128
#define KT 64

__global__ __launch_bounds__(256) void ssb_attn_kernel(
    const bf16* __restrict__ qbuf, const bf16* __restrict__ kbuf,
    const bf16* __restrict__ qkv, bf16* __restrict__ cat)
{
    __shared__ bf16 Ks[KT][HD_];       // 16 KB, B-frag ready for QK^T
    __shared__ bf16 Vt[HD_][72];       // ~18 KB, V transposed, B-frag ready for PV
    __shared__ bf16 Ps[8][16][64];     // 16 KB, per-wave P staging

    const int h    = blockIdx.y;
    const int q0   = blockIdx.x * QT;
    const int tid  = threadIdx.x;
    const int wid  = tid >> 5;
    const int lane = tid & 31;
    const int half = lane >> 4;
    const int l15  = lane & 15;

    // Q fragments for this wave's 16 rows, K=128 -> 4 k-steps, direct from global
    Frag16 aq[4];
    {
        const bf16* qrow = qbuf + ((size_t)h * L_ + q0 + wid * 16 + l15) * HD_;
        for (int ks = 0; ks < 4; ++ks) {
            aq[ks].h[0] = *(const v8bf*)&qrow[ks * 32 + 8 * half];
            aq[ks].h[1] = *(const v8bf*)&qrow[ks * 32 + 16 + 8 * half];
        }
    }

    v8f   o[8] = {};
    float mi[8], li[8];
    for (int r = 0; r < 8; ++r) { mi[r] = -3.0e38f; li[r] = 0.f; }

    const float scl = 0.08838834764831845f;   // 1/sqrt(128)

    for (int kt = 0; kt < L_ / KT; ++kt) {
        // ---- K tile 64x128 row-major (async copy if available)
        for (int i = tid; i < KT * HD_ / 8; i += 256) {
            int r = i >> 4, c = (i & 15) * 8;
            copy16_lds(&kbuf[((size_t)h * L_ + kt * KT + r) * HD_ + c], &Ks[r][c]);
        }
        // ---- V tile 64x128, transposed into Vt[d][key]
        {
            int r  = tid >> 5;
            int d0 = lane * 4;
            for (int p = 0; p < 8; ++p) {
                int key = p * 8 + r;
                const bf16* src =
                    qkv + (size_t)(kt * KT + key) * NQKV_ + 2 * HID_ + h * HD_ + d0;
                Vt[d0][key] = src[0]; Vt[d0 + 1][key] = src[1];
                Vt[d0 + 2][key] = src[2]; Vt[d0 + 3][key] = src[3];
            }
        }
        async_join();
        __syncthreads();

        // ---- S = Q K^T : 4 N-tiles x 4 K-steps
        v8f s[4] = {};
        for (int nt = 0; nt < 4; ++nt) {
            int col = nt * 16 + l15;
            for (int ks = 0; ks < 4; ++ks) {
                Frag16 b;
                b.h[0] = *(const v8bf*)&Ks[col][ks * 32 + 8 * half];
                b.h[1] = *(const v8bf*)&Ks[col][ks * 32 + 16 + 8 * half];
                s[nt] = wmma_bf16(aq[ks], b, s[nt]);
            }
        }

        // ---- online softmax per row (row = r + 8*half, spread over 16 lanes)
        float alpha[8];
        for (int r = 0; r < 8; ++r) {
            float mx = -3.0e38f;
            for (int nt = 0; nt < 4; ++nt) { s[nt][r] *= scl; mx = fmaxf(mx, s[nt][r]); }
            for (int msk = 1; msk < 16; msk <<= 1) mx = fmaxf(mx, __shfl_xor(mx, msk));
            float mnew = fmaxf(mi[r], mx);
            float a = __expf(mi[r] - mnew);
            float rs = 0.f;
            for (int nt = 0; nt < 4; ++nt) {
                float p = __expf(s[nt][r] - mnew);
                s[nt][r] = p; rs += p;
            }
            for (int msk = 1; msk < 16; msk <<= 1) rs += __shfl_xor(rs, msk);
            li[r] = li[r] * a + rs;
            mi[r] = mnew;
            alpha[r] = a;
        }
        for (int nh = 0; nh < 8; ++nh)
            for (int r = 0; r < 8; ++r) o[nh][r] *= alpha[r];

        // ---- stage P (bf16) per-wave, A-frag friendly row-major [16][64]
        for (int nt = 0; nt < 4; ++nt)
            for (int r = 0; r < 8; ++r)
                Ps[wid][r + 8 * half][nt * 16 + l15] = (bf16)s[nt][r];

        // ---- O += P V : 2 K-steps (64 keys) x 8 hd-tiles
        for (int ks = 0; ks < 2; ++ks) {
            Frag16 pa;
            pa.h[0] = *(const v8bf*)&Ps[wid][l15][ks * 32 + 8 * half];
            pa.h[1] = *(const v8bf*)&Ps[wid][l15][ks * 32 + 16 + 8 * half];
            for (int nh = 0; nh < 8; ++nh) {
                Frag16 vb;
                int col = nh * 16 + l15;
                vb.h[0] = *(const v8bf*)&Vt[col][ks * 32 + 8 * half];
                vb.h[1] = *(const v8bf*)&Vt[col][ks * 32 + 16 + 8 * half];
                o[nh] = wmma_bf16(pa, vb, o[nh]);
            }
        }
        __syncthreads();
    }

    // ---- normalize + write into concat buffer (row-major L x NCAT)
    float linv[8];
    for (int r = 0; r < 8; ++r) linv[r] = 1.0f / li[r];
    for (int nh = 0; nh < 8; ++nh) {
        int col = h * HD_ + nh * 16 + l15;
        for (int r = 0; r < 8; ++r) {
            int grow = q0 + wid * 16 + r + 8 * half;
            cat[(size_t)grow * NCAT_ + col] = (bf16)(o[nh][r] * linv[r]);
        }
    }
}

// =========================== launcher =======================================
extern "C" void kernel_launch(void* const* d_in, const int* in_sizes, int n_in,
                              void* d_out, int out_size, void* d_ws, size_t ws_size,
                              hipStream_t stream)
{
    const float* x       = (const float*)d_in[0];
    const float* vec     = (const float*)d_in[1];
    const float* pe      = (const float*)d_in[2];
    const float* mod_w   = (const float*)d_in[3];
    const float* mod_b   = (const float*)d_in[4];
    const float* lin1_w  = (const float*)d_in[5];
    const float* lin1_b  = (const float*)d_in[6];
    const float* lin2_w  = (const float*)d_in[7];
    const float* lin2_b  = (const float*)d_in[8];
    const float* q_scale = (const float*)d_in[9];
    const float* k_scale = (const float*)d_in[10];
    float* out = (float*)d_out;

    // workspace carve-up (256B aligned)
    char* ws = (char*)d_ws;
    size_t off = 0;
    auto take = [&](size_t bytes) {
        void* p = ws + off;
        off = (off + bytes + 255) & ~(size_t)255;
        return p;
    };
    float* m_vec = (float*)take((size_t)NQKV_ * 4);
    bf16*  xmod  = (bf16*) take((size_t)L_ * HID_ * 2);
    bf16*  qkvb  = (bf16*) take((size_t)L_ * NQKV_ * 2);
    bf16*  qbuf  = (bf16*) take((size_t)NH_ * L_ * HD_ * 2);
    bf16*  kbuf  = (bf16*) take((size_t)NH_ * L_ * HD_ * 2);
    bf16*  cat   = (bf16*) take((size_t)L_ * NCAT_ * 2);
    (void)ws_size; (void)in_sizes; (void)n_in; (void)out_size;

    ssb_mod_kernel<<<NQKV_ / 256, 256, 0, stream>>>(vec, mod_w, mod_b, m_vec);

    ssb_lnmod_kernel<<<L_, 256, 0, stream>>>(x, m_vec, xmod);

    // grid: x = M panels (16, concurrent -> weight panel L2 reuse), y = N panels
    ssb_gemm_kernel<0><<<dim3(L_ / BM, N1_ / BN), 256, 0, stream>>>(
        xmod, lin1_w, lin1_b, HID_, N1_, qkvb, cat, nullptr, nullptr, nullptr);

    ssb_qkprep_kernel<<<NH_ * L_ / 8, 256, 0, stream>>>(
        qkvb, pe, q_scale, k_scale, qbuf, kbuf);

    ssb_attn_kernel<<<dim3(L_ / QT, NH_), 256, 0, stream>>>(qbuf, kbuf, qkvb, cat);

    ssb_gemm_kernel<1><<<dim3(L_ / BM, HID_ / BN), 256, 0, stream>>>(
        cat, lin2_w, lin2_b, NCAT_, HID_, nullptr, nullptr, x, m_vec, out);
}
